// QWenAttention_14568529068075
// MI455X (gfx1250) — matrix operand order repaired and checked by reading, not compiled, not run on hardware
//
#include <hip/hip_runtime.h>
#include <hip/hip_bf16.h>

// ---------------------------------------------------------------------------
// Problem constants (from reference)
// ---------------------------------------------------------------------------
#define BB 2
#define SS 2048
#define HID 4096
#define NH 32
#define HD 128
#define QKVN 12288   // 3*HID

typedef __attribute__((ext_vector_type(16))) __bf16 v16bf;
typedef __attribute__((ext_vector_type(16))) short  v16s;
typedef __attribute__((ext_vector_type(8)))  short  v8s;
typedef __attribute__((ext_vector_type(8)))  float  v8f;

typedef __attribute__((ext_vector_type(4))) unsigned int tdm_g0_t;
typedef __attribute__((ext_vector_type(8))) int          tdm_g1_t;
typedef __attribute__((ext_vector_type(4))) int          tdm_g2_t;

__device__ __forceinline__ unsigned short f2bf(float f) {
    unsigned u = __builtin_bit_cast(unsigned, f);
    unsigned r = (u + 0x7FFFu + ((u >> 16) & 1u)) >> 16;   // round-to-nearest-even
    return (unsigned short)r;
}
__device__ __forceinline__ float bf2f(unsigned short s) {
    unsigned u = ((unsigned)s) << 16;
    return __builtin_bit_cast(float, u);
}

__device__ __forceinline__ v8f wmma_bf16(v16s a, v16s b, v8f c) {
    return __builtin_amdgcn_wmma_f32_16x16x32_bf16(
        false, __builtin_bit_cast(v16bf, a),
        false, __builtin_bit_cast(v16bf, b),
        (short)0, c, false, false);
}

// ---------------------------------------------------------------------------
// Tensor Data Mover: 2D bf16 tile (tileW elems wide, tileH rows, row stride
// rowStrideElems) from global -> packed row-major LDS at lds_addr.
// D# layout per CDNA5 ISA; groups 2/3 zero (2D).
// ---------------------------------------------------------------------------
__device__ __forceinline__ void tdm_load_2d_bf16(unsigned lds_addr,
                                                 const unsigned short* gptr,
                                                 unsigned tileW, unsigned tileH,
                                                 unsigned long long rowStrideElems) {
    unsigned long long ga = (unsigned long long)(uintptr_t)gptr;

    tdm_g0_t g0;
    g0[0] = 1u;                                            // count=1 (valid user D#)
    g0[1] = lds_addr;                                      // LDS byte address
    g0[2] = (unsigned)(ga & 0xFFFFFFFFull);                // global_addr[31:0]
    g0[3] = (unsigned)((ga >> 32) & 0x01FFFFFFull)         // global_addr[56:32]
          | (2u << 30);                                    // type=2 ("image")

    const unsigned td0 = (unsigned)rowStrideElems;         // tensor_dim0 (elems)
    const unsigned td1 = 1u << 20;                         // tensor_dim1 (rows, generous)
    const unsigned long long s0 = rowStrideElems;          // tensor_dim0_stride

    tdm_g1_t g1;
    g1[0] = (int)(1u << 16);                               // wg_mask=0, data_size=1 (2B)
    g1[1] = (int)((td0 & 0xFFFFu) << 16);                  // [63:48] = tensor_dim0 lo16
    g1[2] = (int)(((td0 >> 16) & 0xFFFFu) | ((td1 & 0xFFFFu) << 16));
    g1[3] = (int)(((td1 >> 16) & 0xFFFFu) | ((tileW & 0xFFFFu) << 16)); // tile_dim0
    g1[4] = (int)(tileH & 0xFFFFu);                        // tile_dim1 ; tile_dim2=0
    g1[5] = (int)(unsigned)(s0 & 0xFFFFFFFFull);           // dim0_stride[31:0]
    g1[6] = (int)(unsigned)((s0 >> 32) & 0xFFFFull);       // dim0_stride[47:32]
    g1[7] = 0;                                             // dim1_stride hi (unused, 2D)

    tdm_g2_t g2 = {0, 0, 0, 0};
    tdm_g2_t g3 = {0, 0, 0, 0};
#if defined(__clang_major__) && (__clang_major__ >= 23)
    tdm_g1_t g4 = {0, 0, 0, 0, 0, 0, 0, 0};
    __builtin_amdgcn_tensor_load_to_lds(g0, g1, g2, g3, g4, 0);
#else
    __builtin_amdgcn_tensor_load_to_lds(g0, g1, g2, g3, 0);
#endif
}

// ---------------------------------------------------------------------------
// Kernel 1: fp32 -> bf16 conversion (grid-stride)
// ---------------------------------------------------------------------------
__global__ void cvt_f32_bf16(const float* __restrict__ in,
                             unsigned short* __restrict__ out, size_t n) {
    size_t i = (size_t)blockIdx.x * blockDim.x + threadIdx.x;
    size_t stride = (size_t)gridDim.x * blockDim.x;
    for (; i < n; i += stride) out[i] = f2bf(in[i]);
}

// ---------------------------------------------------------------------------
// Kernel 2/5: bf16 GEMM  C[M,N] = A[M,K] * B[K,N] (+bias), WMMA bf16->f32
// Block: 256 threads (8 waves). Block tile 128x128, wave tile 64x32.
// A tile staged by the Tensor Data Mover (wave 0); B staged transposed by ALU.
// ---------------------------------------------------------------------------
template <bool OUT_BF16>
__global__ __launch_bounds__(256)
void gemm_bf16_kernel(const unsigned short* __restrict__ A,
                      const unsigned short* __restrict__ Bm,
                      const float* __restrict__ bias,
                      void* __restrict__ Cout,
                      int M, int N, int K) {
    __shared__ __attribute__((aligned(64))) unsigned short aS[128 * 32]; // [m][k]
    __shared__ __attribute__((aligned(64))) unsigned short bS[128 * 32]; // [n][k] (transposed)
    (void)M;

    const int tid  = threadIdx.x;
    const int wave = tid >> 5;
    const int lane = tid & 31;
    const int h    = lane >> 4;     // half-wave select
    const int ln   = lane & 15;
    const int mBase = blockIdx.y * 128;
    const int nBase = blockIdx.x * 128;
    const int wm = (wave >> 2) * 64;  // 2 waves along M
    const int wn = (wave & 3) * 32;   // 4 waves along N

    const unsigned aS_lds = (unsigned)(uintptr_t)(void*)aS;

    v8f acc[4][2];
    for (int i = 0; i < 4; i++)
        for (int j = 0; j < 2; j++) acc[i][j] = {};

    const int bK    = tid & 31;
    const int bNoff = (tid >> 5) * 16;

    for (int k0 = 0; k0 < K; k0 += 32) {
        // --- stage A tile via TDM: 128 rows x 32 k, row stride K, packed [m][k]
        if (wave == 0)
            tdm_load_2d_bf16(aS_lds, A + (size_t)mBase * K + k0,
                             /*tileW=*/32, /*tileH=*/128, (unsigned long long)K);
        // --- stage B tile transposed: [n][k] so fragment reads are contiguous
        {
            const unsigned short* g = Bm + (size_t)(k0 + bK) * N + nBase + bNoff;
            v16s v = *(const v16s*)g;
            for (int i = 0; i < 16; i++) bS[(bNoff + i) * 32 + bK] = (unsigned short)v[i];
        }
        if (k0 + 32 < K)
            __builtin_prefetch(Bm + (size_t)(k0 + 32 + bK) * N + nBase + bNoff, 0, 0);

        __builtin_amdgcn_s_wait_tensorcnt(0);   // wave0's DMA done (no-op elsewhere)
        __syncthreads();

        // --- fragments: A(16x32 bf16): elems 0-7 k=h*8+e, 8-15 k=16+h*8+e
        v16s aF[4], bF[2];
        for (int i = 0; i < 4; i++) {
            const unsigned short* base = aS + (wm + i * 16 + ln) * 32;
            v8s lo = *(const v8s*)(base + h * 8);
            v8s hi = *(const v8s*)(base + 16 + h * 8);
            v16s t;
            for (int e = 0; e < 8; e++) { t[e] = lo[e]; t[e + 8] = hi[e]; }
            aF[i] = t;
        }
        // --- B(32x16 bf16): lane col = ln, elems 0-7 k=h*8+e, 8-15 k=16+h*8+e
        for (int j = 0; j < 2; j++) {
            const unsigned short* bb = bS + (wn + j * 16 + ln) * 32;
            v8s lo = *(const v8s*)(bb + h * 8);
            v8s hi = *(const v8s*)(bb + 16 + h * 8);
            v16s t;
            for (int e = 0; e < 8; e++) { t[e] = lo[e]; t[e + 8] = hi[e]; }
            bF[j] = t;
        }

        for (int i = 0; i < 4; i++)
            for (int j = 0; j < 2; j++)
                acc[i][j] = wmma_bf16(aF[i], bF[j], acc[i][j]);
        __syncthreads();
    }

    // --- epilogue: C/D layout row = v + 8*h, col = ln
    for (int i = 0; i < 4; i++) {
        for (int j = 0; j < 2; j++) {
            const int n = nBase + wn + j * 16 + ln;
            const float bv = bias ? bias[n] : 0.0f;
            for (int v = 0; v < 8; v++) {
                const int m = mBase + wm + i * 16 + h * 8 + v;
                const float val = acc[i][j][v] + bv;
                if (OUT_BF16)
                    ((unsigned short*)Cout)[(size_t)m * N + n] = f2bf(val);
                else
                    ((float*)Cout)[(size_t)m * N + n] = val;
            }
        }
    }
}

// ---------------------------------------------------------------------------
// Kernel 3: RoPE applied in place on bf16 qkv buffer (q and k sections)
// ---------------------------------------------------------------------------
__global__ void rope_kernel(unsigned short* __restrict__ qkv,
                            const int* __restrict__ positions) {
    const int idx = blockIdx.x * blockDim.x + threadIdx.x;   // < 2^23
    const int d2   = idx & 63;
    const int head = (idx >> 6) & 31;
    const int s    = (idx >> 11) & (SS - 1);
    const int b    = idx >> 22;
    const int pos  = positions[b * SS + s];

    const float inv = __expf(-((float)(2 * d2) / (float)HD) * logf(10000.0f));
    const float ang = (float)pos * inv;
    const float c = cosf(ang), sn = sinf(ang);

    size_t base = ((size_t)b * SS + s) * QKVN + head * HD;
    {
        float x1 = bf2f(qkv[base + d2]);
        float x2 = bf2f(qkv[base + 64 + d2]);
        qkv[base + d2]      = f2bf(x1 * c - x2 * sn);
        qkv[base + 64 + d2] = f2bf(x2 * c + x1 * sn);
    }
    {
        size_t kb = base + HID;
        float x1 = bf2f(qkv[kb + d2]);
        float x2 = bf2f(qkv[kb + 64 + d2]);
        qkv[kb + d2]      = f2bf(x1 * c - x2 * sn);
        qkv[kb + 64 + d2] = f2bf(x2 * c + x1 * sn);
    }
}

// ---------------------------------------------------------------------------
// Kernel 4: causal flash attention, bf16 WMMA, fp32 online softmax
// grid: (S/64, B*NH), block: 128 (4 waves); each wave owns 16 q rows.
// K tile staged by TDM (wave 0); V tile transposed by ALU scatter.
// ---------------------------------------------------------------------------
__global__ __launch_bounds__(128)
void attn_kernel(const unsigned short* __restrict__ qkv,
                 unsigned short* __restrict__ ctx) {
    __shared__ __attribute__((aligned(64))) unsigned short Kt[32 * 128];  // [key][d]
    __shared__ __attribute__((aligned(64))) unsigned short Vt[128 * 32];  // [d][key]
    __shared__ __attribute__((aligned(64))) unsigned short Pb[4][16 * 32];

    const int tid  = threadIdx.x;
    const int wave = tid >> 5;
    const int lane = tid & 31;
    const int h    = lane >> 4;
    const int ln   = lane & 15;

    const int qTile = blockIdx.x;          // 0..31
    const int bh    = blockIdx.y;          // 0..63
    const int b     = bh >> 5;
    const int head  = bh & 31;
    const int qRow0 = qTile * 64 + wave * 16;

    const size_t rs = QKVN;
    const unsigned short* base = qkv + (size_t)b * SS * rs + head * HD;
    const unsigned Kt_lds = (unsigned)(uintptr_t)(void*)Kt;

    // --- Q fragments, 16 rows x 128 d, 4 chunks of K=32
    v16s qF[4];
    for (int c = 0; c < 4; c++) {
        const unsigned short* p = base + (size_t)(qRow0 + ln) * rs + c * 32 + h * 8;
        v8s lo = *(const v8s*)p;
        v8s hi = *(const v8s*)(p + 16);
        v16s t;
        for (int e = 0; e < 8; e++) { t[e] = lo[e]; t[e + 8] = hi[e]; }
        qF[c] = t;
    }

    v8f oAcc[8];
    for (int jd = 0; jd < 8; jd++) oAcc[jd] = {};
    float mRun[8], lRun[8];
    for (int v = 0; v < 8; v++) { mRun[v] = -1e30f; lRun[v] = 0.0f; }
    const float scale = 0.08838834764831845f;  // 1/sqrt(128)

    const int ktEnd = qTile * 2 + 2;          // uniform per block (causal bound)
    for (int kt = 0; kt < ktEnd; kt++) {
        const int key0 = kt * 32;
        // --- stage K tile via TDM: 32 rows x 128 d, stride QKVN, packed [key][d]
        if (wave == 0)
            tdm_load_2d_bf16(Kt_lds, base + (size_t)key0 * rs + HID,
                             /*tileW=*/128, /*tileH=*/32, (unsigned long long)rs);
        // --- stage V tile transposed [d][key]
        {
            const int key = tid & 31;
            const int dB  = (tid >> 5) * 32;
            const unsigned short* g = base + (size_t)(key0 + key) * rs + 2 * HID + dB;
            v16s v0 = *(const v16s*)g;
            v16s v1 = *(const v16s*)(g + 16);
            for (int i = 0; i < 16; i++) {
                Vt[(dB + i) * 32 + key]      = (unsigned short)v0[i];
                Vt[(dB + 16 + i) * 32 + key] = (unsigned short)v1[i];
            }
        }
        __builtin_amdgcn_s_wait_tensorcnt(0);   // DMA done (no-op for waves 1-3)
        __syncthreads();

        // --- scores: 16 q rows x 32 keys (two 16-col WMMA tiles)
        v8f sc[2];
        for (int j = 0; j < 2; j++) {
            v8f c = {};
            for (int cc = 0; cc < 4; cc++) {
                const unsigned short* kp = Kt + (j * 16 + ln) * 128 + cc * 32 + h * 8;
                v8s klo = *(const v8s*)kp;
                v8s khi = *(const v8s*)(kp + 16);
                v16s bF;
                for (int e = 0; e < 8; e++) { bF[e] = klo[e]; bF[e + 8] = khi[e]; }
                c = wmma_bf16(qF[cc], bF, c);
            }
            sc[j] = c;
        }

        // --- causal mask + online softmax
        for (int v = 0; v < 8; v++) {
            const int q  = qRow0 + h * 8 + v;
            const int kA = key0 + ln;
            const int kB = key0 + 16 + ln;
            float s0 = sc[0][v] * scale; if (kA > q) s0 = -1e30f;
            float s1 = sc[1][v] * scale; if (kB > q) s1 = -1e30f;

            float mx = fmaxf(s0, s1);
            for (int off = 1; off < 16; off <<= 1) mx = fmaxf(mx, __shfl_xor(mx, off, 32));
            const float mNew  = fmaxf(mRun[v], mx);
            const float alpha = __expf(mRun[v] - mNew);
            const float p0 = __expf(s0 - mNew);
            const float p1 = __expf(s1 - mNew);
            sc[0][v] = p0; sc[1][v] = p1;
            float ps = p0 + p1;
            for (int off = 1; off < 16; off <<= 1) ps += __shfl_xor(ps, off, 32);
            lRun[v] = lRun[v] * alpha + ps;
            mRun[v] = mNew;
            for (int jd = 0; jd < 8; jd++) oAcc[jd][v] *= alpha;
        }

        // --- write P (16x32 bf16) to this wave's LDS region
        unsigned short* P = &Pb[wave][0];
        for (int v = 0; v < 8; v++) {
            const int m = h * 8 + v;
            P[m * 32 + ln]      = f2bf(sc[0][v]);
            P[m * 32 + 16 + ln] = f2bf(sc[1][v]);
        }
        __syncthreads();

        // --- P as A-fragment (16x32), then PV: 8 WMMA over d tiles
        v16s pF;
        {
            const unsigned short* pr = P + ln * 32;
            v8s lo = *(const v8s*)(pr + h * 8);
            v8s hi = *(const v8s*)(pr + 16 + h * 8);
            v16s t;
            for (int e = 0; e < 8; e++) { t[e] = lo[e]; t[e + 8] = hi[e]; }
            pF = t;
        }
        for (int jd = 0; jd < 8; jd++) {
            const unsigned short* vp = Vt + (jd * 16 + ln) * 32 + h * 8;
            v8s vlo = *(const v8s*)vp;
            v8s vhi = *(const v8s*)(vp + 16);
            v16s vF;
            for (int e = 0; e < 8; e++) { vF[e] = vlo[e]; vF[e + 8] = vhi[e]; }
            oAcc[jd] = wmma_bf16(pF, vF, oAcc[jd]);
        }
        __syncthreads();
    }

    // --- epilogue: ctx[b, q, head*128 + d] bf16
    for (int jd = 0; jd < 8; jd++) {
        for (int v = 0; v < 8; v++) {
            const int q = qRow0 + h * 8 + v;
            const int d = jd * 16 + ln;
            const float val = oAcc[jd][v] / lRun[v];
            ctx[((size_t)b * SS + q) * HID + head * HD + d] = f2bf(val);
        }
    }
}

// ---------------------------------------------------------------------------
// Host launcher
// ---------------------------------------------------------------------------
extern "C" void kernel_launch(void* const* d_in, const int* in_sizes, int n_in,
                              void* d_out, int out_size, void* d_ws, size_t ws_size,
                              hipStream_t stream) {
    (void)in_sizes; (void)n_in; (void)out_size; (void)ws_size;

    const float* hid  = (const float*)d_in[0];
    const int*   pos  = (const int*)d_in[1];
    const float* wqkv = (const float*)d_in[2];
    const float* bqkv = (const float*)d_in[3];
    const float* wo   = (const float*)d_in[4];
    float* out = (float*)d_out;

    const size_t nHid  = (size_t)BB * SS * HID;        // 16,777,216
    const size_t nWqkv = (size_t)HID * QKVN;           // 50,331,648
    const size_t nWo   = (size_t)HID * HID;            // 16,777,216
    const size_t nQkv  = (size_t)BB * SS * QKVN;       // 50,331,648

    unsigned short* hidBF  = (unsigned short*)d_ws;
    unsigned short* wqkvBF = hidBF + nHid;
    unsigned short* woBF   = wqkvBF + nWqkv;
    unsigned short* qkvBF  = woBF + nWo;
    unsigned short* ctxBF  = qkvBF + nQkv;

    // 1) fp32 -> bf16 conversions
    cvt_f32_bf16<<<4096, 256, 0, stream>>>(hid,  hidBF,  nHid);
    cvt_f32_bf16<<<8192, 256, 0, stream>>>(wqkv, wqkvBF, nWqkv);
    cvt_f32_bf16<<<4096, 256, 0, stream>>>(wo,   woBF,   nWo);

    // 2) QKV projection + bias: [4096,4096] x [4096,12288] -> bf16
    gemm_bf16_kernel<true><<<dim3(QKVN / 128, (BB * SS) / 128), 256, 0, stream>>>(
        hidBF, wqkvBF, bqkv, qkvBF, BB * SS, QKVN, HID);

    // 3) RoPE in place on q,k
    rope_kernel<<<(BB * SS * NH * 64) / 256, 256, 0, stream>>>(qkvBF, pos);

    // 4) causal attention -> ctx bf16 [B,S,HID]
    attn_kernel<<<dim3(SS / 64, BB * NH), 128, 0, stream>>>(qkvBF, ctxBF);

    // 5) output projection: [4096,4096] x [4096,4096] -> fp32 d_out
    gemm_bf16_kernel<false><<<dim3(HID / 128, (BB * SS) / 128), 256, 0, stream>>>(
        ctxBF, woBF, nullptr, out, BB * SS, HID, HID);
}
